// Model_20547123544449
// MI455X (gfx1250) — compile-verified, matrix operand order checked by reading, not run on hardware
//
#include <hip/hip_runtime.h>

// ---------------------------------------------------------------------------
// 6-layer GRU forward for MI455X (gfx1250), wave32 + WMMA bf16 + TDM.
//   S=1024, B=128, IN=128, H=256, 3H=768, LAYERS=6
// Per layer:
//   kernel 1: gi = X @ W_ih^T + b_ih          (parallel WMMA GEMM, LDS-staged)
//   kernel 2: persistent scan over t:         (8 WGs; W_hh held in VGPRs,
//             staged via tensor_load_to_lds)
//       gh = h @ W_hh^T + b_hh  (WMMA bf16)
//       r,z = sigmoid; n = tanh; h = (1-z)*n + z*h
// ---------------------------------------------------------------------------

#define S_LEN   1024
#define BATCH   128
#define HID     256
#define G3      768          // 3*HID
#define NWG_SCAN 8

typedef __bf16 bf16;
typedef __bf16 v16bf __attribute__((ext_vector_type(16)));
typedef __bf16 v8bf  __attribute__((ext_vector_type(8)));
typedef float  v8f   __attribute__((ext_vector_type(8)));
typedef unsigned u32x4 __attribute__((ext_vector_type(4)));
typedef unsigned u32x8 __attribute__((ext_vector_type(8)));

// -------------------------------------------------------------------------
// gi GEMM:  gi[M=131072, 768] = X[M, Din] * W[768, Din]^T + bias
// 128x128 C-block per WG (256 thr = 8 waves); K staged in LDS as bf16.
// -------------------------------------------------------------------------
__global__ __launch_bounds__(256)
void gi_gemm_kernel(const float* __restrict__ X, const float* __restrict__ W,
                    const float* __restrict__ bias, float* __restrict__ gi,
                    int Din) {
    __shared__ __align__(16) bf16 lA[128][40];   // 32 K + pad
    __shared__ __align__(16) bf16 lB[128][40];

    const int tid  = threadIdx.x;
    const int lane = tid & 31;
    const int wave = tid >> 5;
    const int row0 = blockIdx.x * 128;           // M block
    const int col0 = blockIdx.y * 128;           // N block (g)

    v8f acc[8];
#pragma unroll
    for (int i = 0; i < 8; ++i) acc[i] = (v8f)0.0f;

    const int r = tid >> 3;                      // 0..31 row group
    const int q = tid & 7;                       // float4 slot in 32-K row

    for (int k0 = 0; k0 < Din; k0 += 32) {
        __syncthreads();
#pragma unroll
        for (int pass = 0; pass < 4; ++pass) {
            const int rr = r + pass * 32;        // 0..127
            float4 a4 = *(const float4*)(X + (size_t)(row0 + rr) * Din + k0 + q * 4);
            lA[rr][q * 4 + 0] = (bf16)a4.x; lA[rr][q * 4 + 1] = (bf16)a4.y;
            lA[rr][q * 4 + 2] = (bf16)a4.z; lA[rr][q * 4 + 3] = (bf16)a4.w;
            float4 b4 = *(const float4*)(W + (size_t)(col0 + rr) * Din + k0 + q * 4);
            lB[rr][q * 4 + 0] = (bf16)b4.x; lB[rr][q * 4 + 1] = (bf16)b4.y;
            lB[rr][q * 4 + 2] = (bf16)b4.z; lB[rr][q * 4 + 3] = (bf16)b4.w;
        }
        __syncthreads();

        // A fragment: lane = row-in-tile; two contiguous 8-K runs per lane
        const int am    = lane & 15;
        const int khalf = (lane >> 4) * 8;
        const bf16* ap  = &lA[wave * 16 + am][khalf];
        v8bf a0 = *(const v8bf*)ap;
        v8bf a1 = *(const v8bf*)(ap + 16);
        v16bf afrag;
#pragma unroll
        for (int i = 0; i < 8; ++i) { afrag[i] = a0[i]; afrag[8 + i] = a1[i]; }

        // load all 8 B fragments first, then run the WMMA chain
        v16bf bfr[8];
#pragma unroll
        for (int nt = 0; nt < 8; ++nt) {
            const bf16* bp = &lB[nt * 16 + am][khalf];
            v8bf b0 = *(const v8bf*)bp;
            v8bf b1 = *(const v8bf*)(bp + 16);
#pragma unroll
            for (int i = 0; i < 8; ++i) { bfr[nt][i] = b0[i]; bfr[nt][8 + i] = b1[i]; }
        }
#pragma unroll
        for (int nt = 0; nt < 8; ++nt) {
            acc[nt] = __builtin_amdgcn_wmma_f32_16x16x32_bf16(
                false, afrag, false, bfr[nt], (short)0, acc[nt], false, false);
        }
    }

    // Epilogue: C/D layout -> lane gives N, vgpr index gives M (+8 for hi lanes)
    const int n  = lane & 15;
    const int hi = lane >> 4;
#pragma unroll
    for (int nt = 0; nt < 8; ++nt) {
        const int g  = col0 + nt * 16 + n;
        const float bv = bias[g];
#pragma unroll
        for (int v = 0; v < 8; ++v) {
            const int m = wave * 16 + v + 8 * hi;
            gi[(size_t)(row0 + m) * G3 + g] = acc[nt][v] + bv;
        }
    }
}

// -------------------------------------------------------------------------
// Persistent GRU scan for one layer. 8 WGs x 256 thr (8 waves, 1 WG/WGP).
// WG wg owns column blocks jb = {2wg, 2wg+1} of H for the three gates
// (6 N-tiles). Wave w owns M-tile w. W_hh slice is staged into LDS by the
// Tensor Data Mover (tensor_load_to_lds) and then held in VGPRs (48 bf16
// fragments) for the entire layer -> zero LDS traffic in the t-loop.
// -------------------------------------------------------------------------
__global__ __launch_bounds__(256, 1)
void gru_scan_kernel(const float* __restrict__ gi, const float* __restrict__ W_hh,
                     const float* __restrict__ b_hh, float* __restrict__ hbuf,
                     float* __restrict__ out, unsigned* __restrict__ counter) {
    // f32 staging for TDM: 3 gates x 32 local g-rows x 256 K = 96 KB
    __shared__ __align__(16) float fW[3][32][HID];

    const int tid  = threadIdx.x;
    const int lane = tid & 31;
    const int wave = tid >> 5;      // m-tile
    const int wg   = blockIdx.x;    // 0..7
    const int am    = lane & 15;
    const int hi    = lane >> 4;
    const int khalf = hi * 8;

    // ---- TDM: DMA each gate's 32x256 f32 W_hh tile into LDS ----
    if (wave == 0) {
#pragma unroll
        for (int gate = 0; gate < 3; ++gate) {
            const unsigned long long ga =
                (unsigned long long)(const void*)(W_hh + ((size_t)gate * HID + (size_t)wg * 32) * HID);
            const unsigned lds_off =
                (unsigned)(unsigned long long)(const void*)&fW[gate][0][0];
            u32x4 g0;
            g0[0] = 1u;                                   // count=1, user mode
            g0[1] = lds_off;                              // lds_addr
            g0[2] = (unsigned)(ga & 0xffffffffu);         // global_addr[31:0]
            g0[3] = (unsigned)((ga >> 32) & 0x01ffffffu)  // global_addr[56:32]
                    | (2u << 30);                         // type=2 (image)
            u32x8 g1;
            g1[0] = 0x00020000u;      // data_size=2 (4 bytes), mask=0
            g1[1] = (256u << 16);     // tensor_dim0 = 256
            g1[2] = (768u << 16);     // tensor_dim1 = 768 (rows remaining OK)
            g1[3] = (256u << 16);     // tile_dim0 = 256
            g1[4] = 32u;              // tile_dim1 = 32
            g1[5] = 256u;             // tensor_dim0_stride = 256
            g1[6] = 0u;
            g1[7] = 0u;
            asm volatile("tensor_load_to_lds %0, %1" :: "s"(g0), "s"(g1) : "memory");
        }
        __builtin_amdgcn_s_wait_tensorcnt(0);
    }
    __syncthreads();

    // ---- build 48 register-resident bf16 B fragments from the f32 LDS ----
    v16bf bw[6][8];                  // [gate*2+jj][k-chunk]
#pragma unroll
    for (int tix = 0; tix < 6; ++tix) {
        const int gate = tix >> 1, jj = tix & 1;
        const float* wp = &fW[gate][jj * 16 + am][0];
#pragma unroll
        for (int kc = 0; kc < 8; ++kc) {
            const float* p = wp + kc * 32 + khalf;
            float4 c0 = *(const float4*)p;
            float4 c1 = *(const float4*)(p + 4);
            float4 c2 = *(const float4*)(p + 16);
            float4 c3 = *(const float4*)(p + 20);
            v16bf f;
            f[0]  = (bf16)c0.x; f[1]  = (bf16)c0.y; f[2]  = (bf16)c0.z; f[3]  = (bf16)c0.w;
            f[4]  = (bf16)c1.x; f[5]  = (bf16)c1.y; f[6]  = (bf16)c1.z; f[7]  = (bf16)c1.w;
            f[8]  = (bf16)c2.x; f[9]  = (bf16)c2.y; f[10] = (bf16)c2.z; f[11] = (bf16)c2.w;
            f[12] = (bf16)c3.x; f[13] = (bf16)c3.y; f[14] = (bf16)c3.z; f[15] = (bf16)c3.w;
            bw[tix][kc] = f;
        }
    }

    // per-lane hidden-bias values for the epilogue (lane gives N column)
    const int nn_ = lane & 15;
    float bias_g[6];
#pragma unroll
    for (int tix = 0; tix < 6; ++tix) {
        const int gate = tix >> 1, jj = tix & 1;
        bias_g[tix] = b_hh[gate * HID + (2 * wg + jj) * 16 + nn_];
    }

    for (int t = 0; t < S_LEN; ++t) {
        const float* hrd = hbuf + (size_t)(t & 1) * (BATCH * HID);
        float*       hwr = hbuf + (size_t)((t + 1) & 1) * (BATCH * HID);
        const float* gi_t = gi + (size_t)t * BATCH * G3;

        // prefetch next timestep's gi rows (global_prefetch_b8)
        if (t + 1 < S_LEN) {
            const float* nx = gi + (size_t)(t + 1) * BATCH * G3 +
                              (size_t)(wave * 16 + am) * G3 + (2 * wg) * 16;
            __builtin_prefetch(nx, 0, 3);
        }

        v8f acc[6];
#pragma unroll
        for (int i = 0; i < 6; ++i) acc[i] = (v8f)0.0f;

#pragma unroll
        for (int kc = 0; kc < 8; ++kc) {
            const int k0 = kc * 32;
            // A fragment: h row (wave*16+am), f32 -> bf16 on the fly
            const float* ap = hrd + (size_t)(wave * 16 + am) * HID + k0 + khalf;
            float4 h0v = *(const float4*)ap;
            float4 h1v = *(const float4*)(ap + 4);
            float4 h2v = *(const float4*)(ap + 16);
            float4 h3v = *(const float4*)(ap + 20);
            v16bf afrag;
            afrag[0]  = (bf16)h0v.x; afrag[1]  = (bf16)h0v.y;
            afrag[2]  = (bf16)h0v.z; afrag[3]  = (bf16)h0v.w;
            afrag[4]  = (bf16)h1v.x; afrag[5]  = (bf16)h1v.y;
            afrag[6]  = (bf16)h1v.z; afrag[7]  = (bf16)h1v.w;
            afrag[8]  = (bf16)h2v.x; afrag[9]  = (bf16)h2v.y;
            afrag[10] = (bf16)h2v.z; afrag[11] = (bf16)h2v.w;
            afrag[12] = (bf16)h3v.x; afrag[13] = (bf16)h3v.y;
            afrag[14] = (bf16)h3v.z; afrag[15] = (bf16)h3v.w;

#pragma unroll
            for (int tix = 0; tix < 6; ++tix) {
                acc[tix] = __builtin_amdgcn_wmma_f32_16x16x32_bf16(
                    false, afrag, false, bw[tix][kc], (short)0, acc[tix], false, false);
            }
        }

        // ---- fused gate epilogue (acc: [jj]=r, [2+jj]=z, [4+jj]=n) ----
#pragma unroll
        for (int jj = 0; jj < 2; ++jj) {
            const int col = (2 * wg + jj) * 16 + nn_;
#pragma unroll
            for (int v = 0; v < 8; ++v) {
                const int b_ = wave * 16 + v + 8 * hi;        // batch row
                const float ghr = acc[0 + jj][v] + bias_g[0 + jj];
                const float ghz = acc[2 + jj][v] + bias_g[2 + jj];
                const float ghn = acc[4 + jj][v] + bias_g[4 + jj];
                const float* gib = gi_t + (size_t)b_ * G3;
                const float gir = gib[0 * HID + col];
                const float giz = gib[1 * HID + col];
                const float gin = gib[2 * HID + col];
                const float rg = 1.0f / (1.0f + __expf(-(gir + ghr)));
                const float zg = 1.0f / (1.0f + __expf(-(giz + ghz)));
                const float ng = tanhf(gin + rg * ghn);
                const float hold = hrd[(size_t)b_ * HID + col];
                const float hnew = (1.0f - zg) * ng + zg * hold;
                hwr[(size_t)b_ * HID + col] = hnew;
                out[((size_t)t * BATCH + b_) * HID + col] = hnew;
            }
        }

        // ---- inter-WG sync for the recurrence ----
        __threadfence();
        __syncthreads();
        if (tid == 0) {
            atomicAdd(counter, 1u);
            const unsigned target = (unsigned)(NWG_SCAN * (t + 1));
            while (atomicAdd(counter, 0u) < target) { __builtin_amdgcn_s_sleep(1); }
        }
        // CDNA5 cluster barrier (s_barrier_signal/-wait -3); NOP outside clusters
        if (wave == 0) __builtin_amdgcn_s_cluster_barrier();
        __syncthreads();
        __threadfence();
    }
}

__global__ void init_counter_kernel(unsigned* c) {
    if (threadIdx.x == 0) *c = 0u;
}

// -------------------------------------------------------------------------
extern "C" void kernel_launch(void* const* d_in, const int* in_sizes, int n_in,
                              void* d_out, int out_size, void* d_ws, size_t ws_size,
                              hipStream_t stream) {
    const float* x         = (const float*)d_in[0];   // (S,B,IN)
    const float* h0        = (const float*)d_in[1];   // (L,B,H)
    const float* W_ih0     = (const float*)d_in[2];   // (3H,IN)
    const float* b_ih0     = (const float*)d_in[3];   // (3H)
    const float* W_ih_rest = (const float*)d_in[4];   // (L-1,3H,H)
    const float* b_ih_rest = (const float*)d_in[5];   // (L-1,3H)
    const float* W_hh      = (const float*)d_in[6];   // (L,3H,H)
    const float* b_hh      = (const float*)d_in[7];   // (L,3H)
    float* out = (float*)d_out;                       // (S,B,H)

    // workspace layout (floats)
    float* gi_buf = (float*)d_ws;                                   // S*B*3H
    float* xbuf   = gi_buf + (size_t)S_LEN * BATCH * G3;            // S*B*H
    float* hbuf   = xbuf   + (size_t)S_LEN * BATCH * HID;           // 2*B*H
    unsigned* counter = (unsigned*)(hbuf + 2 * BATCH * HID);

    const int M = S_LEN * BATCH;   // 131072
    for (int l = 0; l < 6; ++l) {
        const float* X   = (l == 0) ? x : xbuf;
        const int    Din = (l == 0) ? 128 : HID;
        const float* Wi  = (l == 0) ? W_ih0 : W_ih_rest + (size_t)(l - 1) * G3 * HID;
        const float* bi  = (l == 0) ? b_ih0 : b_ih_rest + (size_t)(l - 1) * G3;

        dim3 grid(M / 128, G3 / 128);
        gi_gemm_kernel<<<grid, 256, 0, stream>>>(X, Wi, bi, gi_buf, Din);

        hipMemcpyAsync(hbuf, h0 + (size_t)l * BATCH * HID,
                       (size_t)BATCH * HID * sizeof(float),
                       hipMemcpyDeviceToDevice, stream);
        init_counter_kernel<<<1, 32, 0, stream>>>(counter);

        float* lout = (l == 5) ? out : xbuf;
        gru_scan_kernel<<<NWG_SCAN, 256, 0, stream>>>(
            gi_buf, W_hh + (size_t)l * G3 * HID, b_hh + (size_t)l * G3,
            hbuf, lout, counter);
    }
}
